// FIDSubsequence_3229815406612
// MI455X (gfx1250) — matrix operand order repaired and checked by reading, not compiled, not run on hardware
//
#include <hip/hip_runtime.h>
#include <hip/hip_bf16.h>
#include <stdint.h>

typedef __attribute__((ext_vector_type(16))) _Float16 v16h;
typedef __attribute__((ext_vector_type(8)))  float    v8f;

#define NW    2048        // windows per sequence
#define NB    4096        // both sequences batched
#define HID   128
#define GATES 384
#define WINL  8
#define FEATD 30
#define TLEN  8196
#define INF_  1.0e10f

// ---------------- WMMA fragment loaders (wave32, ISA 7.12.2 layouts) -------

// A-matrix 16x32 f16: lanes 0-15 row M=lane hold K {0..7,16..23};
// lanes 16-31 row M=lane-16 hold K {8..15,24..31}.
__device__ __forceinline__ v16h load_A16(const _Float16* __restrict__ A,
                                         int lda, int m0, int k0) {
  const int lane = threadIdx.x & 31;
  const _Float16* p = A + (size_t)(m0 + (lane & 15)) * lda + k0 + ((lane >> 4) << 3);
  v16h a;
#pragma unroll
  for (int i = 0; i < 8; ++i) { a[i] = p[i]; a[8 + i] = p[16 + i]; }
  return a;
}

// B-matrix 32x16 f16 built as W^T: B[k][n] = W[n][k] (W row-major, ld=ldw).
// Lane n = n0 + lane%16; lanes 0-15 K {0..7,16..23}, lanes 16-31 K {8..15,24..31}.
__device__ __forceinline__ v16h load_BT16(const _Float16* __restrict__ W,
                                          int ldw, int n0, int k0) {
  const int lane = threadIdx.x & 31;
  const _Float16* p = W + (size_t)(n0 + (lane & 15)) * ldw + k0 + ((lane >> 4) << 3);
  v16h b;
#pragma unroll
  for (int i = 0; i < 8; ++i) { b[i] = p[i]; b[8 + i] = p[16 + i]; }
  return b;
}

__device__ __forceinline__ v8f wmma_f32_f16(v16h a, v16h b, v8f c) {
  return __builtin_amdgcn_wmma_f32_16x16x32_f16(false, a, false, b, (short)0, c,
                                                false, false);
}

// ---------------- small helper kernels -------------------------------------

__global__ void conv_f16_kernel(_Float16* __restrict__ dst,
                                const float* __restrict__ src, int n) {
  int i = blockIdx.x * blockDim.x + threadIdx.x;
  if (i < n) dst[i] = (_Float16)src[i];
}

// Wih1 (384,72) -> (384,96) zero-padded f16
__global__ void pad_wih1_kernel(const float* __restrict__ src,
                                _Float16* __restrict__ dst) {
  int i = blockIdx.x * blockDim.x + threadIdx.x;
  if (i >= 384 * 96) return;
  int r = i / 96, c = i - r * 96;
  dst[i] = (c < 72) ? (_Float16)src[r * 72 + c] : (_Float16)0.0f;
}

__global__ void zero_f16_kernel(_Float16* __restrict__ p, int n) {
  int i = blockIdx.x * blockDim.x + threadIdx.x;
  if (i < n) p[i] = (_Float16)0.0f;
}

// Window extraction: Xf[t][n][f] = src[f*TLEN + 4n + t], f<72 (pad 72..95 = 0).
// n in [0,2048) -> x sequence, [2048,4096) -> y sequence.
__global__ void window_kernel(const float* __restrict__ x,
                              const float* __restrict__ y,
                              _Float16* __restrict__ Xf) {
  const int KX = 96;
  int idx = blockIdx.x * blockDim.x + threadIdx.x;
  int total = WINL * NB * KX;
  if (idx >= total) return;
  int t = idx / (NB * KX);
  int r = idx - t * (NB * KX);
  int n = r / KX;
  int f = r - n * KX;
  _Float16 val = (_Float16)0.0f;
  if (f < 72) {
    const float* src = (n < NW) ? x : y;
    int nn = n & (NW - 1);
    val = (_Float16)src[(size_t)f * TLEN + nn * 4 + t];
  }
  Xf[idx] = val;
}

// ---------------- fused GRU step (gi GEMM + gh GEMM + gates), WMMA ---------
// Grid: NB/16 blocks of 256 threads (8 waves). Wave w owns h-columns
// [16w,16w+16): it computes gate N-tiles at columns 16w, 16w+128, 16w+256,
// so r/z/n for its columns land in its own C fragments.
__global__ __launch_bounds__(256) void gru_step_kernel(
    const _Float16* __restrict__ Xt, int ldx, int kxChunks,
    const _Float16* __restrict__ Wih,   // (384, ldx) f16
    const _Float16* __restrict__ Whh,   // (384, 128) f16
    const float* __restrict__ bih, const float* __restrict__ bhh,
    _Float16* __restrict__ h,           // (NB,128) f16, updated in place
    _Float16* __restrict__ hstore)      // optional copy of h_new (layer-1 out)
{
  const int m0  = blockIdx.x * 16;
  const int w   = threadIdx.x >> 5;
  const int hc0 = w * 16;

  // gi = x_t @ Wih^T  (K = 96 or 128)
  v8f giR = {}, giZ = {}, giN = {};
  for (int kc = 0; kc < kxChunks; ++kc) {
    int k0 = kc * 32;
    v16h a = load_A16(Xt, ldx, m0, k0);
    giR = wmma_f32_f16(a, load_BT16(Wih, ldx, hc0,       k0), giR);
    giZ = wmma_f32_f16(a, load_BT16(Wih, ldx, hc0 + 128, k0), giZ);
    giN = wmma_f32_f16(a, load_BT16(Wih, ldx, hc0 + 256, k0), giN);
  }
  // gh = h @ Whh^T  (K = 128)
  v8f ghR = {}, ghZ = {}, ghN = {};
#pragma unroll
  for (int kc = 0; kc < 4; ++kc) {
    int k0 = kc * 32;
    v16h a = load_A16(h, HID, m0, k0);
    ghR = wmma_f32_f16(a, load_BT16(Whh, HID, hc0,       k0), ghR);
    ghZ = wmma_f32_f16(a, load_BT16(Whh, HID, hc0 + 128, k0), ghZ);
    ghN = wmma_f32_f16(a, load_BT16(Whh, HID, hc0 + 256, k0), ghN);
  }
  __syncthreads();  // all waves done reading h rows m0..m0+15 -> safe in-place

  const int lane = threadIdx.x & 31;
  const int col  = hc0 + (lane & 15);
  const int roff = (lane >> 4) << 3;
  const float biR = bih[col], biZ = bih[col + 128], biN = bih[col + 256];
  const float bhR = bhh[col], bhZ = bhh[col + 128], bhN = bhh[col + 256];
#pragma unroll
  for (int v = 0; v < 8; ++v) {
    int m = m0 + v + roff;
    float r  = 1.0f / (1.0f + __expf(-(giR[v] + biR + ghR[v] + bhR)));
    float z  = 1.0f / (1.0f + __expf(-(giZ[v] + biZ + ghZ[v] + bhZ)));
    float nn = tanhf(giN[v] + biN + r * (ghN[v] + bhN));
    float hp = (float)h[(size_t)m * HID + col];
    float hn = (1.0f - z) * nn + z * hp;
    h[(size_t)m * HID + col] = (_Float16)hn;
    if (hstore) hstore[(size_t)m * HID + col] = (_Float16)hn;
  }
}

// ---------------- final projection: feats = h2 @ W1^T + b1 -----------------
__global__ void proj_kernel(const _Float16* __restrict__ h,
                            const float* __restrict__ W1,
                            const float* __restrict__ b1,
                            float* __restrict__ feats) {
  int idx = blockIdx.x * blockDim.x + threadIdx.x;
  if (idx >= NB * FEATD) return;
  int n = idx / FEATD, f = idx - n * FEATD;
  float s = b1[f];
  const _Float16* hr = h + (size_t)n * HID;
  const float* wr = W1 + (size_t)f * HID;
#pragma unroll 8
  for (int k = 0; k < HID; ++k) s += (float)hr[k] * wr[k];
  feats[idx] = s;
}

// ---------------- pairwise squared distance --------------------------------
__global__ void pairwise_kernel(const float* __restrict__ feats,
                                float* __restrict__ D) {
  int j = blockIdx.x * 16 + threadIdx.x;
  int i = blockIdx.y * 16 + threadIdx.y;
  const float* f1 = feats + (size_t)i * FEATD;
  const float* f2 = feats + (size_t)(NW + j) * FEATD;
  float s = 0.0f;
#pragma unroll
  for (int t = 0; t < FEATD; ++t) { float d = f1[t] - f2[t]; s += d * d; }
  D[(size_t)i * NW + j] = s;
}

// ---------------- soft-DTW anti-diagonal scan (single persistent WG) -------
__global__ __launch_bounds__(1024) void softdtw_kernel(
    const float* __restrict__ D, float* __restrict__ out) {
  __shared__ float A2[NW], A1[NW], AN[NW];
  const int tid = threadIdx.x;
  for (int i = tid; i < NW; i += 1024) { A2[i] = INF_; A1[i] = INF_; }
  __syncthreads();
  for (int k = 0; k < 2 * NW - 1; ++k) {
#pragma unroll
    for (int ii = 0; ii < 2; ++ii) {
      int i = tid + ii * 1024;
      int j = k - i;
      bool valid = (j >= 0) && (j < NW);
      int jc = j < 0 ? 0 : (j > NW - 1 ? NW - 1 : j);
      float d    = D[(size_t)i * NW + jc];
      float diag = (i == 0) ? ((k == 0) ? 0.0f : INF_) : A2[i - 1];
      float up   = (i == 0) ? INF_ : A1[i - 1];
      float cur  = A1[i];
      float mn = fminf(diag, fminf(up, cur));
      float s;
      if (mn >= 0.5f * INF_) s = INF_;
      else s = mn - __logf(__expf(mn - diag) + __expf(mn - up) + __expf(mn - cur));
      AN[i] = valid ? (d + s) : INF_;
    }
    __syncthreads();
#pragma unroll
    for (int ii = 0; ii < 2; ++ii) {
      int i = tid + ii * 1024;
      A2[i] = A1[i]; A1[i] = AN[i];
    }
    __syncthreads();
  }
  if (tid == 0) out[0] = A1[NW - 1];
}

// ---------------- launcher --------------------------------------------------
extern "C" void kernel_launch(void* const* d_in, const int* in_sizes, int n_in,
                              void* d_out, int out_size, void* d_ws, size_t ws_size,
                              hipStream_t stream) {
  (void)in_sizes; (void)n_in; (void)out_size; (void)ws_size;
  const float* x_xyz = (const float*)d_in[0];
  const float* y_xyz = (const float*)d_in[1];
  const float* Wih1  = (const float*)d_in[2];
  const float* Whh1  = (const float*)d_in[3];
  const float* bih1  = (const float*)d_in[4];
  const float* bhh1  = (const float*)d_in[5];
  const float* Wih2  = (const float*)d_in[6];
  const float* Whh2  = (const float*)d_in[7];
  const float* bih2  = (const float*)d_in[8];
  const float* bhh2  = (const float*)d_in[9];
  const float* W1    = (const float*)d_in[10];
  const float* b1    = (const float*)d_in[11];
  float* out = (float*)d_out;

  char* p = (char*)d_ws;
  auto alloc = [&p](size_t bytes) -> char* {
    char* r = p; p += (bytes + 255) & ~(size_t)255; return r;
  };
  _Float16* Xf16  = (_Float16*)alloc((size_t)WINL * NB * 96 * 2);   // 6.3 MB
  _Float16* h1buf = (_Float16*)alloc((size_t)WINL * NB * HID * 2);  // 8.4 MB
  _Float16* hcur  = (_Float16*)alloc((size_t)NB * HID * 2);         // 1.0 MB
  _Float16* Wih1h = (_Float16*)alloc((size_t)GATES * 96 * 2);
  _Float16* Whh1h = (_Float16*)alloc((size_t)GATES * HID * 2);
  _Float16* Wih2h = (_Float16*)alloc((size_t)GATES * HID * 2);
  _Float16* Whh2h = (_Float16*)alloc((size_t)GATES * HID * 2);
  float*    feats = (float*)alloc((size_t)NB * FEATD * 4);
  float*    Dmat  = (float*)alloc((size_t)NW * NW * 4);             // 16.8 MB

  // Weight conversion / padding (f32 -> f16)
  pad_wih1_kernel<<<(384 * 96 + 255) / 256, 256, 0, stream>>>(Wih1, Wih1h);
  conv_f16_kernel<<<(GATES * HID + 255) / 256, 256, 0, stream>>>(Whh1h, Whh1, GATES * HID);
  conv_f16_kernel<<<(GATES * HID + 255) / 256, 256, 0, stream>>>(Wih2h, Wih2, GATES * HID);
  conv_f16_kernel<<<(GATES * HID + 255) / 256, 256, 0, stream>>>(Whh2h, Whh2, GATES * HID);

  // Window extraction for both sequences
  {
    int total = WINL * NB * 96;
    window_kernel<<<(total + 255) / 256, 256, 0, stream>>>(x_xyz, y_xyz, Xf16);
  }

  // Layer 1: h starts at zero; 8 recurrent steps, store per-step h into h1buf
  zero_f16_kernel<<<(NB * HID + 255) / 256, 256, 0, stream>>>(hcur, NB * HID);
  for (int t = 0; t < WINL; ++t) {
    gru_step_kernel<<<NB / 16, 256, 0, stream>>>(
        Xf16 + (size_t)t * NB * 96, 96, 3, Wih1h, Whh1h, bih1, bhh1,
        hcur, h1buf + (size_t)t * NB * HID);
  }

  // Layer 2: h starts at zero; input is layer-1 outputs; only final h needed
  zero_f16_kernel<<<(NB * HID + 255) / 256, 256, 0, stream>>>(hcur, NB * HID);
  for (int t = 0; t < WINL; ++t) {
    gru_step_kernel<<<NB / 16, 256, 0, stream>>>(
        h1buf + (size_t)t * NB * HID, HID, 4, Wih2h, Whh2h, bih2, bhh2,
        hcur, (_Float16*)nullptr);
  }

  // Final projection, pairwise distances, soft-DTW
  proj_kernel<<<(NB * FEATD + 255) / 256, 256, 0, stream>>>(hcur, W1, b1, feats);
  pairwise_kernel<<<dim3(NW / 16, NW / 16), dim3(16, 16), 0, stream>>>(feats, Dmat);
  softdtw_kernel<<<1, 1024, 0, stream>>>(Dmat, out);
}
